// CausalSelfAttention_26087631356641
// MI455X (gfx1250) — matrix operand order repaired and checked by reading, not compiled, not run on hardware
//
#include <hip/hip_runtime.h>
#include <hip/hip_bf16.h>
#include <math.h>

#define BATCH 2
#define SEQ   2048
#define EMBD  1024
#define NHEAD 16
#define HD    64
#define QKVN  (3*EMBD)

typedef __attribute__((ext_vector_type(16))) __bf16 v16bf;
typedef __attribute__((ext_vector_type(8)))  float  v8f;

union Frag { v16bf v; unsigned int u[8]; };

// ---- fragment helpers (layouts per CDNA5 ISA 7.12.2) --------------------

// A matrix 16x32 bf16, row-major source, leading dim ld (elements).
__device__ __forceinline__ v16bf load_a_frag(const __bf16* p, int ld) {
  const int lane = threadIdx.x & 31;
  const int row  = lane & 15;
  const int kb   = (lane >> 4) << 3;          // 0 or 8
  Frag f;
  const unsigned short* q = (const unsigned short*)p + (size_t)row * ld;
#pragma unroll
  for (int j = 0; j < 8; ++j) {
    int k = ((j >> 2) << 4) + kb + ((j & 3) << 1);
    f.u[j] = *(const unsigned int*)(q + k);   // (k, k+1) contiguous
  }
  return f.v;
}

// B matrix 32x16 bf16 where element (k,n) lives at pT[n*ldt + k]
// (column-major storage, so K pairs are contiguous dwords).
__device__ __forceinline__ v16bf load_bt_frag(const __bf16* pT, int ldt) {
  const int lane = threadIdx.x & 31;
  const int col  = lane & 15;
  const int kb   = (lane >> 4) << 3;          // 0 or 8
  Frag f;
  const unsigned short* q = (const unsigned short*)pT + (size_t)col * ldt + kb;
#pragma unroll
  for (int j = 0; j < 8; ++j)
    f.u[j] = *(const unsigned int*)(q + ((j >> 2) << 4) + ((j & 3) << 1));
  return f.v;
}

__device__ __forceinline__ v8f wmma_bf16(v16bf a, v16bf b, v8f c) {
  return __builtin_amdgcn_wmma_f32_16x16x32_bf16(
      false, a, false, b, (short)0, c, false, false);
}

// ---- conversion kernels -------------------------------------------------

__global__ void cvt_bf16(const float* __restrict__ in, __bf16* __restrict__ out, int n) {
  int i = blockIdx.x * blockDim.x + threadIdx.x;
  if (i < n) out[i] = (__bf16)in[i];
}

// in: [K][N] f32 row-major -> out: [N][K] bf16 ; LDS-tiled, coalesced both sides.
// grid = (N/32, K/32), block = (32, 8). K, N multiples of 32.
__global__ void cvt_bf16_t(const float* __restrict__ in, __bf16* __restrict__ out,
                           int K, int N) {
  __shared__ float tile[32][33];
  int nb = blockIdx.x * 32, kb = blockIdx.y * 32;
  int tx = threadIdx.x, ty = threadIdx.y;
#pragma unroll
  for (int i = 0; i < 32; i += 8)
    tile[ty + i][tx] = in[(size_t)(kb + ty + i) * N + nb + tx];
  __syncthreads();
#pragma unroll
  for (int i = 0; i < 32; i += 8)
    out[(size_t)(nb + ty + i) * K + kb + tx] = (__bf16)tile[tx][ty + i];
}

// ---- QKV GEMM: [4096,1024] x [1024,3072] ; 16x64 strip per wave ---------
// A fragment loaded once per K-step, reused across 4 WMMAs.
// Writes Q pre-scaled by 0.125*log2(e) so attention can use exp2f.
__global__ void qkv_gemm(const __bf16* __restrict__ X, const __bf16* __restrict__ WT,
                         __bf16* __restrict__ Q, __bf16* __restrict__ K,
                         __bf16* __restrict__ Vt) {
  const int NT4 = QKVN / 64;                         // 48 strips in N
  int wid = (blockIdx.x * blockDim.x + threadIdx.x) >> 5;
  int mt = wid / NT4, n4 = wid - mt * NT4;
  const int lane = threadIdx.x & 31;

  const __bf16* a = X  + (size_t)mt * 16 * EMBD;
  const __bf16* b = WT + (size_t)n4 * 64 * EMBD;
  v8f c0 = {}, c1 = {}, c2 = {}, c3 = {};
  for (int kk = 0; kk < EMBD; kk += 32) {
    __builtin_prefetch(a + kk + 64, 0, 1);
    v16bf af = load_a_frag(a + kk, EMBD);
    c0 = wmma_bf16(af, load_bt_frag(b + kk,                EMBD), c0);
    c1 = wmma_bf16(af, load_bt_frag(b + 16 * EMBD + kk,    EMBD), c1);
    c2 = wmma_bf16(af, load_bt_frag(b + 32 * EMBD + kk,    EMBD), c2);
    c3 = wmma_bf16(af, load_bt_frag(b + 48 * EMBD + kk,    EMBD), c3);
  }

  const int col = lane & 15, rb = (lane >> 4) << 3;
  // 64-col strip never straddles q/k/v or head boundaries (64 | 1024).
  int n0 = n4 * 64;
  int which = n0 / EMBD;
  int cc = n0 - which * EMBD;
  int h = cc / HD;
  const float qscale = 0.18033688011112042f;         // 0.125 * log2(e)
  v8f cts[4] = {c0, c1, c2, c3};
#pragma unroll
  for (int tt = 0; tt < 4; ++tt) {
    int d = tt * 16 + col;
#pragma unroll
    for (int r = 0; r < 8; ++r) {
      int m = mt * 16 + rb + r;
      int bi = m / SEQ, t = m - bi * SEQ;
      float v = cts[tt][r];
      if (which == 0)
        Q[(((size_t)bi * NHEAD + h) * SEQ + t) * HD + d] = (__bf16)(v * qscale);
      else if (which == 1)
        K[(((size_t)bi * NHEAD + h) * SEQ + t) * HD + d] = (__bf16)v;
      else
        Vt[(((size_t)bi * NHEAD + h) * HD + d) * SEQ + t] = (__bf16)v;
    }
  }
}

// ---- fused causal flash attention: one wave = 16 query rows -------------
__global__ void attn_fused(const __bf16* __restrict__ Q, const __bf16* __restrict__ K,
                           const __bf16* __restrict__ Vt, __bf16* __restrict__ Y) {
  __shared__ __bf16 pbuf[8][16 * 32];                // per-wave P staging (1 KB each)
  const int wib = threadIdx.x >> 5;
  int wid = blockIdx.x * 8 + wib;
  int bh = wid >> 7;                                 // / (SEQ/16)
  int qb = wid & 127;
  int q0 = qb * 16;
  const int lane = threadIdx.x & 31;
  const int col = lane & 15, rb = (lane >> 4) << 3;

  const __bf16* Qp = Q  + (size_t)bh * SEQ * HD;
  const __bf16* Kp = K  + (size_t)bh * SEQ * HD;
  const __bf16* Vp = Vt + (size_t)bh * HD * SEQ;

  v16bf aq0 = load_a_frag(Qp + (size_t)q0 * HD,      HD);   // d = 0..31
  v16bf aq1 = load_a_frag(Qp + (size_t)q0 * HD + 32, HD);   // d = 32..63

  v8f o0 = {}, o1 = {}, o2 = {}, o3 = {};
  float mrow[8], lrow[8];
#pragma unroll
  for (int r = 0; r < 8; ++r) { mrow[r] = -__builtin_inff(); lrow[r] = 0.0f; }

  __bf16* pb = &pbuf[wib][0];
  int nkb = (q0 + 47) >> 5;                          // 32-key blocks covering <= q0+15
  for (int kb = 0; kb < nkb; ++kb) {
    int k0 = kb * 32;
    v8f s0 = {}, s1 = {};
    s0 = wmma_bf16(aq0, load_bt_frag(Kp + (size_t)k0 * HD,             HD), s0);
    s0 = wmma_bf16(aq1, load_bt_frag(Kp + (size_t)k0 * HD + 32,        HD), s0);
    s1 = wmma_bf16(aq0, load_bt_frag(Kp + (size_t)(k0 + 16) * HD,      HD), s1);
    s1 = wmma_bf16(aq1, load_bt_frag(Kp + (size_t)(k0 + 16) * HD + 32, HD), s1);

#pragma unroll
    for (int r = 0; r < 8; ++r) {
      int qidx = q0 + rb + r;
      float v0 = (k0 + col      <= qidx) ? s0[r] : -__builtin_inff();
      float v1 = (k0 + 16 + col <= qidx) ? s1[r] : -__builtin_inff();
      float bm = fmaxf(v0, v1);
#pragma unroll
      for (int off = 1; off < 16; off <<= 1)         // 16-lane row reduction
        bm = fmaxf(bm, __shfl_xor(bm, off, 32));
      float newm = fmaxf(mrow[r], bm);
      float sc = exp2f(mrow[r] - newm);
      float p0 = exp2f(v0 - newm);
      float p1 = exp2f(v1 - newm);
      float ps = p0 + p1;
#pragma unroll
      for (int off = 1; off < 16; off <<= 1)
        ps += __shfl_xor(ps, off, 32);
      lrow[r] = lrow[r] * sc + ps;
      mrow[r] = newm;
      o0[r] *= sc; o1[r] *= sc; o2[r] *= sc; o3[r] *= sc;
      int prow = rb + r;
      pb[prow * 32 + col]      = (__bf16)p0;
      pb[prow * 32 + col + 16] = (__bf16)p1;
    }
    asm volatile("s_wait_dscnt 0" ::: "memory");     // LDS in-order per wave; cross-lane RAW
    v16bf ap = load_a_frag(pb, 32);                  // P as 16x32 A fragment
    o0 = wmma_bf16(ap, load_bt_frag(Vp + (size_t)0 * 16 * SEQ + k0, SEQ), o0);
    o1 = wmma_bf16(ap, load_bt_frag(Vp + (size_t)1 * 16 * SEQ + k0, SEQ), o1);
    o2 = wmma_bf16(ap, load_bt_frag(Vp + (size_t)2 * 16 * SEQ + k0, SEQ), o2);
    o3 = wmma_bf16(ap, load_bt_frag(Vp + (size_t)3 * 16 * SEQ + k0, SEQ), o3);
  }

  int b = bh >> 4, h = bh & 15;
#pragma unroll
  for (int r = 0; r < 8; ++r) {
    float inv = 1.0f / lrow[r];
    int t = q0 + rb + r;
    __bf16* yp = Y + ((size_t)b * SEQ + t) * EMBD + h * HD;
    yp[ 0 + col] = (__bf16)(o0[r] * inv);
    yp[16 + col] = (__bf16)(o1[r] * inv);
    yp[32 + col] = (__bf16)(o2[r] * inv);
    yp[48 + col] = (__bf16)(o3[r] * inv);
  }
}

// ---- output projection: [4096,1024] x [1024,1024], 16x64 strip per wave -
__global__ void proj_gemm(const __bf16* __restrict__ Yb, const __bf16* __restrict__ WT,
                          float* __restrict__ Out) {
  const int NT4 = EMBD / 64;                         // 16 strips
  int wid = (blockIdx.x * blockDim.x + threadIdx.x) >> 5;
  int mt = wid / NT4, n4 = wid - mt * NT4;
  const int lane = threadIdx.x & 31;

  const __bf16* a = Yb + (size_t)mt * 16 * EMBD;
  const __bf16* b = WT + (size_t)n4 * 64 * EMBD;
  v8f c0 = {}, c1 = {}, c2 = {}, c3 = {};
  for (int kk = 0; kk < EMBD; kk += 32) {
    __builtin_prefetch(a + kk + 64, 0, 1);
    v16bf af = load_a_frag(a + kk, EMBD);
    c0 = wmma_bf16(af, load_bt_frag(b + kk,             EMBD), c0);
    c1 = wmma_bf16(af, load_bt_frag(b + 16 * EMBD + kk, EMBD), c1);
    c2 = wmma_bf16(af, load_bt_frag(b + 32 * EMBD + kk, EMBD), c2);
    c3 = wmma_bf16(af, load_bt_frag(b + 48 * EMBD + kk, EMBD), c3);
  }
  const int col = lane & 15, rb = (lane >> 4) << 3;
  v8f cts[4] = {c0, c1, c2, c3};
#pragma unroll
  for (int tt = 0; tt < 4; ++tt)
#pragma unroll
    for (int r = 0; r < 8; ++r)
      Out[(size_t)(mt * 16 + rb + r) * EMBD + n4 * 64 + tt * 16 + col] = cts[tt][r];
}

// ---- host launcher ------------------------------------------------------

extern "C" void kernel_launch(void* const* d_in, const int* in_sizes, int n_in,
                              void* d_out, int out_size, void* d_ws, size_t ws_size,
                              hipStream_t stream) {
  const float* x  = (const float*)d_in[0];    // [2,2048,1024]
  const float* Wa = (const float*)d_in[1];    // [1024,3072]
  const float* Wp = (const float*)d_in[2];    // [1024,1024]
  float* out = (float*)d_out;                 // [2,2048,1024] f32

  char* ws = (char*)d_ws;
  size_t off = 0;
  __bf16* Xb  = (__bf16*)(ws + off); off += (size_t)BATCH * SEQ * EMBD * 2;        // 8 MB
  __bf16* WaT = (__bf16*)(ws + off); off += (size_t)QKVN * EMBD * 2;               // 6 MB
  __bf16* WpT = (__bf16*)(ws + off); off += (size_t)EMBD * EMBD * 2;               // 2 MB
  __bf16* Qb  = (__bf16*)(ws + off); off += (size_t)BATCH * NHEAD * SEQ * HD * 2;  // 8 MB
  __bf16* Kb  = (__bf16*)(ws + off); off += (size_t)BATCH * NHEAD * SEQ * HD * 2;  // 8 MB
  __bf16* Vt  = (__bf16*)(ws + off); off += (size_t)BATCH * NHEAD * HD * SEQ * 2;  // 8 MB
  __bf16* Yb  = (__bf16*)(ws + off); off += (size_t)BATCH * SEQ * EMBD * 2;        // 8 MB

  const int TPB = 256;

  int nx = BATCH * SEQ * EMBD;
  cvt_bf16<<<(nx + TPB - 1) / TPB, TPB, 0, stream>>>(x, Xb, nx);

  cvt_bf16_t<<<dim3(QKVN / 32, EMBD / 32), dim3(32, 8), 0, stream>>>(Wa, WaT, EMBD, QKVN);
  cvt_bf16_t<<<dim3(EMBD / 32, EMBD / 32), dim3(32, 8), 0, stream>>>(Wp, WpT, EMBD, EMBD);

  // QKV: 256 M-tiles x 48 N-strips, 1 strip per wave, 8 waves per block
  int qkv_waves = (BATCH * SEQ / 16) * (QKVN / 64);  // 12288
  qkv_gemm<<<qkv_waves / 8, TPB, 0, stream>>>(Xb, WaT, Qb, Kb, Vt);

  // attention: 2*16*(2048/16) = 4096 wave tasks
  int attn_waves = BATCH * NHEAD * (SEQ / 16);
  attn_fused<<<attn_waves / 8, TPB, 0, stream>>>(Qb, Kb, Vt, Yb);

  // projection: 256 M-tiles x 16 N-strips
  int proj_waves = (BATCH * SEQ / 16) * (EMBD / 64); // 4096
  proj_gemm<<<proj_waves / 8, TPB, 0, stream>>>(Yb, WpT, out);
}
